// V_FNO_long_38671885533652
// MI455X (gfx1250) — matrix operand order repaired and checked by reading, not compiled, not run on hardware
//
#include <hip/hip_runtime.h>
#include <math.h>

// ---------------- problem constants ----------------
#define B_    1024
#define SX_   180
#define SY_   130
#define C_    3
#define MM_   16          // modes per dim
#define NMODE_ 32         // 2*MM_
#define OUT_  180
#define PLANE_ (SX_*SY_)          // 23400
#define FLAT_  (C_*PLANE_)        // 70200
#define FLATP_ 70208              // FLAT_ padded to K-tile multiple (32)
#define HID1_  2808
#define HID1P_ 2816               // padded N / K
#define W2ROWS_ 256               // out1_w2 rows padded (>= 2 N-tiles)
#define HID1KT_ 2194              // 70208/32
#define HID2KT_ 88                // 2816/32

typedef __bf16 bf16;
typedef __attribute__((ext_vector_type(16))) bf16  v16bf;
typedef __attribute__((ext_vector_type(8)))  float v8f;

// async global->LDS path (guarded; host pass & missing-builtin fall back)
// clang's prototype takes pointers to 'int __attribute__((vector_size(16)))'
#if defined(__HIP_DEVICE_COMPILE__) && __has_builtin(__builtin_amdgcn_global_load_async_to_lds_b128)
#define HAS_ASYNC_LDS 1
typedef int gv4i __attribute__((vector_size(16)));
#define GLB_V4(p) ((__attribute__((address_space(1))) gv4i*)(p))
#define LDS_V4(p) ((__attribute__((address_space(3))) gv4i*)(p))
#if __has_builtin(__builtin_amdgcn_s_wait_asynccnt)
#define ASYNC_WAIT_N(n) __builtin_amdgcn_s_wait_asynccnt(n)
#else
#define ASYNC_WAIT_N(n) asm volatile("s_wait_asynccnt %0" :: "i"(n) : "memory")
#endif
#else
#define HAS_ASYNC_LDS 0
#define ASYNC_WAIT_N(n)
#endif

// f32 -> bf16 round-to-nearest-even
__device__ __forceinline__ unsigned short f2bf(float f) {
  union { float f; unsigned u; } x; x.f = f;
  unsigned r = x.u + 0x7FFFu + ((x.u >> 16) & 1u);
  return (unsigned short)(r >> 16);
}
__device__ __forceinline__ unsigned pack2bf(float a, float b) {
  return (unsigned)f2bf(a) | ((unsigned)f2bf(b) << 16);
}

// =====================================================================
// Guarded f32-input WMMA GEMM (used for the small DFT stages).
// C[m,n] = sum_k A[m,k]*W[n,k]; A (MxK) f32, W (NxK) f32.
// Kreal must be even (true: 130, 32); pairs never straddle the K guard.
// =====================================================================
__global__ __launch_bounds__(256) void k_gemm(
    const float* __restrict__ A, int lda,
    const float* __restrict__ W, int ldw,
    float* __restrict__ Cmat, int ldc,
    int Nreal, int Kreal, int Ktiles)
{
  __shared__ unsigned short As[128 * 32];
  __shared__ unsigned short Bs[128 * 32];
  const int t    = threadIdx.x;
  const int lane = t & 31;
  const int w    = t >> 5;
  const int wm   = w >> 2;
  const int wn   = w & 3;
  const int hf   = lane >> 4;
  const int r    = lane & 15;
  const int m0   = blockIdx.y * 128;
  const int n0   = blockIdx.x * 128;

  const v8f vz = {0.f,0.f,0.f,0.f,0.f,0.f,0.f,0.f};
  v8f acc[4][2];
  #pragma unroll
  for (int i = 0; i < 4; ++i)
    #pragma unroll
    for (int j = 0; j < 2; ++j) acc[i][j] = vz;

  for (int kt = 0; kt < Ktiles; ++kt) {
    const int k0 = kt * 32;
    __syncthreads();
    // pair-wise staging: b64 load, 2 converts, packed b32 LDS store
    #pragma unroll
    for (int v = 0; v < 8; ++v) {
      int idx = v * 256 + t;           // pair index 0..2047
      int mm  = idx >> 4;              // row 0..127
      int kk  = (idx & 15) * 2;        // even col
      int kg  = k0 + kk;
      unsigned pa = 0u, pb = 0u;
      if (kg < Kreal) {
        const float* ap = A + (size_t)(m0 + mm) * lda + kg;
        pa = pack2bf(ap[0], ap[1]);
      }
      int ng = n0 + mm;
      if (ng < Nreal && kg < Kreal) {
        const float* bp = W + (size_t)ng * ldw + kg;
        pb = pack2bf(bp[0], bp[1]);
      }
      *(unsigned*)&As[mm * 32 + kk] = pa;
      *(unsigned*)&Bs[mm * 32 + kk] = pb;
    }
    __syncthreads();

    union { unsigned short u[16]; v16bf v; } afr[4], bfr[2];
    #pragma unroll
    for (int fm = 0; fm < 4; ++fm) {
      int ml = wm * 64 + fm * 16 + r;
      #pragma unroll
      for (int e = 0; e < 8; ++e) {
        afr[fm].u[e]     = As[ml * 32 + hf * 16 + e];
        afr[fm].u[8 + e] = As[ml * 32 + hf * 16 + 8 + e];
      }
    }
    #pragma unroll
    for (int fn = 0; fn < 2; ++fn) {
      int nl = wn * 32 + fn * 16 + r;
      #pragma unroll
      for (int e = 0; e < 16; ++e)
        bfr[fn].u[e] = Bs[nl * 32 + hf * 16 + e];
    }
    #pragma unroll
    for (int fm = 0; fm < 4; ++fm)
      #pragma unroll
      for (int fn = 0; fn < 2; ++fn)
        acc[fm][fn] = __builtin_amdgcn_wmma_f32_16x16x32_bf16(
            false, afr[fm].v, false, bfr[fn].v,
            (short)0, acc[fm][fn], false, false);
  }

  #pragma unroll
  for (int fm = 0; fm < 4; ++fm)
    #pragma unroll
    for (int fn = 0; fn < 2; ++fn) {
      int n = n0 + wn * 32 + fn * 16 + r;
      if (n >= Nreal) continue;
      #pragma unroll
      for (int rr = 0; rr < 8; ++rr) {
        int m = m0 + wm * 64 + fm * 16 + hf * 8 + rr;
        Cmat[(size_t)m * ldc + n] = acc[fm][fn][rr];
      }
    }
}

// =====================================================================
// Guard-free bf16-operand WMMA GEMM (MLP stages), LDS DOUBLE-BUFFERED:
// tile k+1 is fetched with global_load_async_to_lds_b128 while tile k's
// 8 WMMAs execute; s_wait_asynccnt 4 retires exactly the previous tile's
// loads (async loads complete in order). Operands padded: no guards.
// =====================================================================
__global__ __launch_bounds__(256) void k_gemm_bf(
    const unsigned short* __restrict__ A, int ldA,
    const unsigned short* __restrict__ W, int ldW,
    unsigned short* __restrict__ outh, int Npad,
    float* __restrict__ outf, int ldc,
    const float* __restrict__ bias,
    int Nreal, int Ktiles, int relu)
{
  __shared__ unsigned short As[2][128 * 32];
  __shared__ unsigned short Bs[2][128 * 32];
  const int t    = threadIdx.x;
  const int lane = t & 31;
  const int w    = t >> 5;
  const int wm   = w >> 2;
  const int wn   = w & 3;
  const int hf   = lane >> 4;
  const int r    = lane & 15;
  const int m0   = blockIdx.y * 128;
  const int n0   = blockIdx.x * 128;

  const v8f vz = {0.f,0.f,0.f,0.f,0.f,0.f,0.f,0.f};
  v8f acc[4][2];
  #pragma unroll
  for (int i = 0; i < 4; ++i)
    #pragma unroll
    for (int j = 0; j < 2; ++j) acc[i][j] = vz;

  // stage one 128x32 bf16 tile of A and of W into LDS buffer `sel`
  auto stage = [&](int kt, int sel) {
    const int k0 = kt * 32;
    #pragma unroll
    for (int v = 0; v < 2; ++v) {
      int idx = v * 256 + t;
      int rr  = idx >> 2;            // tile row
      int cc  = (idx & 3) * 8;       // 8 bf16 = 16B chunk
      const unsigned short* ga = A + (size_t)(m0 + rr) * ldA + k0 + cc;
      const unsigned short* gb = W + (size_t)(n0 + rr) * ldW + k0 + cc;
#if HAS_ASYNC_LDS
      __builtin_amdgcn_global_load_async_to_lds_b128(
          GLB_V4(ga), LDS_V4(&As[sel][rr * 32 + cc]), 0, 0);
      __builtin_amdgcn_global_load_async_to_lds_b128(
          GLB_V4(gb), LDS_V4(&Bs[sel][rr * 32 + cc]), 0, 0);
#else
      *(uint4*)&As[sel][rr * 32 + cc] = *(const uint4*)ga;
      *(uint4*)&Bs[sel][rr * 32 + cc] = *(const uint4*)gb;
#endif
    }
  };

  stage(0, 0);                       // prologue: tile 0 in flight
  for (int kt = 0; kt < Ktiles; ++kt) {
    const int cur = kt & 1;
    if (kt + 1 < Ktiles) {
      stage(kt + 1, cur ^ 1);        // prefetch next tile (other buffer)
      ASYNC_WAIT_N(4);               // retire current tile's 4 loads
    } else {
      ASYNC_WAIT_N(0);
    }
    __syncthreads();                 // current tile visible to all waves

    const unsigned short* Ac = As[cur];
    const unsigned short* Bc = Bs[cur];
    union { unsigned short u[16]; v16bf v; } afr[4], bfr[2];
    #pragma unroll
    for (int fm = 0; fm < 4; ++fm) {
      int ml = wm * 64 + fm * 16 + r;
      #pragma unroll
      for (int e = 0; e < 8; ++e) {
        afr[fm].u[e]     = Ac[ml * 32 + hf * 16 + e];
        afr[fm].u[8 + e] = Ac[ml * 32 + hf * 16 + 8 + e];
      }
    }
    #pragma unroll
    for (int fn = 0; fn < 2; ++fn) {
      int nl = wn * 32 + fn * 16 + r;
      #pragma unroll
      for (int e = 0; e < 16; ++e)
        bfr[fn].u[e] = Bc[nl * 32 + hf * 16 + e];
    }
    #pragma unroll
    for (int fm = 0; fm < 4; ++fm)
      #pragma unroll
      for (int fn = 0; fn < 2; ++fn)
        acc[fm][fn] = __builtin_amdgcn_wmma_f32_16x16x32_bf16(
            false, afr[fm].v, false, bfr[fn].v,
            (short)0, acc[fm][fn], false, false);
    __syncthreads();                 // buffer safe to overwrite next iter
  }

  #pragma unroll
  for (int fm = 0; fm < 4; ++fm)
    #pragma unroll
    for (int fn = 0; fn < 2; ++fn) {
      int n = n0 + wn * 32 + fn * 16 + r;
      float bv = (bias && n < Nreal) ? bias[n] : 0.f;
      #pragma unroll
      for (int rr = 0; rr < 8; ++rr) {
        int m = m0 + wm * 64 + fm * 16 + hf * 8 + rr;
        float v = acc[fm][fn][rr] + bv;
        if (relu) v = fmaxf(v, 0.f);
        if (outh) {
          if (n < Npad)
            outh[(size_t)m * ldc + n] = (n < Nreal) ? f2bf(v) : (unsigned short)0;
        } else if (n < Nreal) {
          outf[(size_t)m * ldc + n] = v;
        }
      }
    }
}

// =====================================================================
// f32 -> padded bf16 matrix convert (rowsSrc x ksrc) -> (rowsDst x kdst)
// pads (extra rows / cols) zero-filled. ksrc, kdst multiples of 8.
// =====================================================================
__global__ __launch_bounds__(256) void k_cvt(
    const float* __restrict__ src, unsigned short* __restrict__ dst,
    int rowsSrc, int ksrc, int kdst, long nchunk)
{
  long id0 = (long)blockIdx.x * blockDim.x + threadIdx.x;
  long stride = (long)gridDim.x * blockDim.x;
  int cpr = kdst >> 3;
  for (long id = id0; id < nchunk; id += stride) {
    int r  = (int)(id / cpr);
    int c8 = (int)(id % cpr) << 3;
    union { unsigned short u[8]; uint4 q; } o;
    if (r < rowsSrc && c8 < ksrc) {
      const float4* s4 = (const float4*)(src + (size_t)r * ksrc + c8);
      float4 a = s4[0], b = s4[1];
      o.u[0] = f2bf(a.x); o.u[1] = f2bf(a.y); o.u[2] = f2bf(a.z); o.u[3] = f2bf(a.w);
      o.u[4] = f2bf(b.x); o.u[5] = f2bf(b.y); o.u[6] = f2bf(b.z); o.u[7] = f2bf(b.w);
    } else {
      o.q = make_uint4(0, 0, 0, 0);
    }
    *(uint4*)&dst[(size_t)r * kdst + c8] = o.q;
  }
}

// zero the 8-column K-pad of the flattened bf16 activations
__global__ void k_zeropad(unsigned short* __restrict__ flath) {
  int i = blockIdx.x * blockDim.x + threadIdx.x;
  if (i < B_ * 8)
    flath[(size_t)(i >> 3) * FLATP_ + FLAT_ + (i & 7)] = 0;
}

// =====================================================================
// Twiddle matrices (recomputed each call; deterministic).
// =====================================================================
__global__ void k_twiddle(float* __restrict__ Fy1, float* __restrict__ Fy2) {
  const float PI2 = 6.283185307179586f;
  for (int i = threadIdx.x; i < 32 * 160; i += blockDim.x) {
    int n = i / 160, k = i % 160;
    int ky = n & 15;
    float v = 0.f;
    if (k < SY_) {
      float ww = PI2 * (float)(ky * k) / (float)SY_;
      v = (n < 16) ? cosf(ww) : -sinf(ww);
    }
    Fy1[i] = v;
  }
  for (int i = threadIdx.x; i < SY_ * 32; i += blockDim.x) {
    int n = i >> 5, k = i & 31;
    int ky = k & 15;
    float ww = PI2 * (float)(ky * n) / (float)SY_;
    float alpha = ((ky == 0) ? 1.f : 2.f) / (float)SY_;
    float v = (k < 16) ? alpha * cosf(ww)
                       : ((ky == 0) ? 0.f : -alpha * sinf(ww));
    Fy2[i] = v;
  }
}

// =====================================================================
// Build channels [value, gx, gy] + instance-norm, one block per (b,c)
// =====================================================================
__device__ __forceinline__ float build_val(const float* __restrict__ x,
                                           int b, int c, int xi, int yi) {
  if (c == 0) return x[((size_t)b * SX_ + xi) * (SY_ + 1) + yi];
  if (c == 1) return (float)xi * (1.f / (SX_ - 1));
  return (float)yi * (1.f / (SY_ - 1));
}

__global__ __launch_bounds__(256) void k_build(const float* __restrict__ x,
                                               float* __restrict__ out) {
  __shared__ float s1[256], s2[256];
  int p = blockIdx.x, b = p / C_, c = p % C_;
  int t = threadIdx.x;
  float sum = 0.f, ss = 0.f;
  for (int i = t; i < PLANE_; i += 256) {
    float v = build_val(x, b, c, i / SY_, i % SY_);
    sum += v; ss += v * v;
  }
  s1[t] = sum; s2[t] = ss; __syncthreads();
  for (int o = 128; o > 0; o >>= 1) {
    if (t < o) { s1[t] += s1[t + o]; s2[t] += s2[t + o]; }
    __syncthreads();
  }
  float mean = s1[0] * (1.f / PLANE_);
  float var  = s2[0] * (1.f / PLANE_) - mean * mean;
  float rstd = rsqrtf(var + 1e-5f);
  for (int i = t; i < PLANE_; i += 256) {
    float v = build_val(x, b, c, i / SY_, i % SY_);
    out[(size_t)p * PLANE_ + i] = (v - mean) * rstd;
  }
}

// =====================================================================
// Instance-norm (+optional exact GELU); optional f32 out, optional
// bf16 out into the padded flattened activation buffer.
// =====================================================================
__global__ __launch_bounds__(256) void k_inorm(const float* __restrict__ in,
                                               float* __restrict__ out,
                                               unsigned short* __restrict__ bfout,
                                               int do_gelu) {
  __shared__ float s1[256], s2[256];
  int p = blockIdx.x, t = threadIdx.x;
  int b = p / C_, c = p % C_;
  const float* src = in + (size_t)p * PLANE_;
  float sum = 0.f, ss = 0.f;
  for (int i = t; i < PLANE_; i += 256) { float v = src[i]; sum += v; ss += v * v; }
  s1[t] = sum; s2[t] = ss; __syncthreads();
  for (int o = 128; o > 0; o >>= 1) {
    if (t < o) { s1[t] += s1[t + o]; s2[t] += s2[t + o]; }
    __syncthreads();
  }
  float mean = s1[0] * (1.f / PLANE_);
  float var  = s2[0] * (1.f / PLANE_) - mean * mean;
  float rstd = rsqrtf(var + 1e-5f);
  for (int i = t; i < PLANE_; i += 256) {
    float v = (src[i] - mean) * rstd;
    if (do_gelu) v = 0.5f * v * (1.f + erff(v * 0.70710678118f));
    if (out)   out[(size_t)p * PLANE_ + i] = v;
    if (bfout) bfout[(size_t)b * FLATP_ + (size_t)c * PLANE_ + i] = f2bf(v);
  }
}

// =====================================================================
// Per-batch middle stage: forward x-DFT (32 kx modes), 3x3 complex
// channel mixing, inverse x-DFT. f32, LDS staged.
// =====================================================================
__global__ __launch_bounds__(256) void k_spectral_mid(
    const float* __restrict__ G, const float* __restrict__ w1,
    const float* __restrict__ w2, float* __restrict__ H)
{
  extern __shared__ float sm[];
  float* sG = sm;                                  // 3*180*32
  float* sX = sG + C_ * SX_ * NMODE_;              // 3*32*16*2
  float* sO = sX + C_ * NMODE_ * MM_ * 2;
  int b = blockIdx.x, t = threadIdx.x;
  const float* Gb = G + (size_t)b * C_ * SX_ * NMODE_;
  for (int i = t; i < C_ * SX_ * NMODE_; i += 256) sG[i] = Gb[i];
  __syncthreads();

  const float W180 = 6.283185307179586f / 180.f;

  for (int id = t; id < C_ * NMODE_ * MM_; id += 256) {
    int c  = id / (NMODE_ * MM_);
    int j  = (id / MM_) % NMODE_;
    int ky = id % MM_;
    int kx = (j < 16) ? j : (148 + j);
    float re = 0.f, im = 0.f;
    for (int xx = 0; xx < SX_; ++xx) {
      float ang = -W180 * (float)(kx * xx);
      float sw, cw; __sincosf(ang, &sw, &cw);
      float gr = sG[(c * SX_ + xx) * NMODE_ + ky];
      float gi = sG[(c * SX_ + xx) * NMODE_ + 16 + ky];
      re += gr * cw - gi * sw;
      im += gr * sw + gi * cw;
    }
    sX[id * 2] = re; sX[id * 2 + 1] = im;
  }
  __syncthreads();

  for (int id = t; id < C_ * NMODE_ * MM_; id += 256) {
    int o  = id / (NMODE_ * MM_);
    int j  = (id / MM_) % NMODE_;
    int ky = id % MM_;
    int jj = j & 15;
    const float* wp = (j < 16) ? w1 : w2;
    float orr = 0.f, oii = 0.f;
    for (int i = 0; i < C_; ++i) {
      const float* wc = wp + ((((size_t)i * C_ + o) * MM_ + jj) * MM_ + ky) * 2;
      float wr = wc[0], wi = wc[1];
      float xr = sX[((i * NMODE_ + j) * MM_ + ky) * 2];
      float xi = sX[((i * NMODE_ + j) * MM_ + ky) * 2 + 1];
      orr += xr * wr - xi * wi;
      oii += xr * wi + xi * wr;
    }
    sO[id * 2] = orr; sO[id * 2 + 1] = oii;
  }
  __syncthreads();

  for (int id = t; id < C_ * SX_ * MM_; id += 256) {
    int c  = id / (SX_ * MM_);
    int xx = (id / MM_) % SX_;
    int ky = id % MM_;
    float hr = 0.f, hi = 0.f;
    for (int j = 0; j < NMODE_; ++j) {
      int kx = (j < 16) ? j : (148 + j);
      float ang = W180 * (float)(kx * xx);
      float sw, cw; __sincosf(ang, &sw, &cw);
      float orr = sO[((c * NMODE_ + j) * MM_ + ky) * 2];
      float oii = sO[((c * NMODE_ + j) * MM_ + ky) * 2 + 1];
      hr += orr * cw - oii * sw;
      hi += orr * sw + oii * cw;
    }
    size_t row = ((size_t)b * C_ + c) * SX_ + xx;
    H[row * NMODE_ + ky]      = hr * (1.f / SX_);
    H[row * NMODE_ + 16 + ky] = hi * (1.f / SX_);
  }
}

// =====================================================================
// Final regression head, one block per batch element.
// =====================================================================
__global__ __launch_bounds__(256) void k_final(
    const float* __restrict__ hout, const float* __restrict__ x,
    const float* __restrict__ w1, const float* __restrict__ b1,
    const float* __restrict__ w2, const float* __restrict__ b2,
    float* __restrict__ out)
{
  __shared__ float vv[2 * OUT_];
  __shared__ float tb[OUT_];
  int b = blockIdx.x, t = threadIdx.x;
  if (t < OUT_) {
    vv[2 * t]     = hout[(size_t)b * OUT_ + t];
    vv[2 * t + 1] = (x[((size_t)b * SX_ + t) * (SY_ + 1) + SY_] - 400.f) * 0.01f;
  }
  __syncthreads();
  if (t < OUT_) {
    float a = b1[t];
    for (int k = 0; k < 2 * OUT_; ++k) a += w1[t * 2 * OUT_ + k] * vv[k];
    tb[t] = fmaxf(a, 0.f);
  }
  __syncthreads();
  if (t < OUT_) {
    float a = b2[t];
    for (int k = 0; k < OUT_; ++k) a += w2[t * OUT_ + k] * tb[k];
    out[(size_t)b * OUT_ + t] = a * 100.f + 400.f;
  }
}

// =====================================================================
extern "C" void kernel_launch(void* const* d_in, const int* in_sizes, int n_in,
                              void* d_out, int out_size, void* d_ws, size_t ws_size,
                              hipStream_t stream) {
  const float* x       = (const float*)d_in[0];
  const float* wA[2]   = {(const float*)d_in[1], (const float*)d_in[3]};
  const float* wB[2]   = {(const float*)d_in[2], (const float*)d_in[4]};
  const float* out1_w1 = (const float*)d_in[5];
  const float* out1_b1 = (const float*)d_in[6];
  const float* out1_w2 = (const float*)d_in[7];
  const float* out1_b2 = (const float*)d_in[8];
  const float* reg2_w1 = (const float*)d_in[9];
  const float* reg2_b1 = (const float*)d_in[10];
  const float* reg2_w2 = (const float*)d_in[11];
  const float* reg2_b2 = (const float*)d_in[12];
  float* outp = (float*)d_out;

  // ---- workspace partition ----
  const size_t NP = (size_t)B_ * C_ * PLANE_;        // 71,884,800 f32
  const size_t NG = (size_t)B_ * C_ * SX_ * NMODE_;  // 17,694,720 f32
  float* bufA = (float*)d_ws;
  float* bufT = bufA + NP;
  float* Gbuf = bufT + NP;
  float* Hbuf = Gbuf + NG;
  float* hout = Hbuf + NG;                    // 1024*180
  float* Fy1  = hout + (size_t)B_ * OUT_;     // 32*160
  float* Fy2  = Fy1 + 32 * 160;               // 130*32
  unsigned short* w1h   = (unsigned short*)(Fy2 + SY_ * 32); // 2816 x 70208
  unsigned short* w2h   = w1h + (size_t)HID1P_ * FLATP_;     // 256 x 2816
  unsigned short* flath = w2h + (size_t)W2ROWS_ * HID1P_;    // 1024 x 70208
  unsigned short* hid1h = flath + (size_t)B_ * FLATP_;       // 1024 x 2816

  const int Mrows = B_ * C_ * SX_;            // 552960 = 4320*128

  hipLaunchKernelGGL(k_twiddle, dim3(1), dim3(256), 0, stream, Fy1, Fy2);
  // pre-convert MLP weights to padded bf16 (deterministic, every call)
  hipLaunchKernelGGL(k_cvt, dim3(4096), dim3(256), 0, stream,
      out1_w1, w1h, HID1_, FLAT_, FLATP_,
      (long)HID1P_ * (FLATP_ / 8));
  hipLaunchKernelGGL(k_cvt, dim3(512), dim3(256), 0, stream,
      out1_w2, w2h, OUT_, HID1_, HID1P_,
      (long)W2ROWS_ * (HID1P_ / 8));
  hipLaunchKernelGGL(k_zeropad, dim3(32), dim3(256), 0, stream, flath);

  hipLaunchKernelGGL(k_build, dim3(B_ * C_), dim3(256), 0, stream, x, bufA);

  const size_t shb = (size_t)(C_ * SX_ * NMODE_ + 2 * C_ * NMODE_ * MM_ * 2) * sizeof(float);
  for (int L = 0; L < 2; ++L) {
    // forward y-DFT (WMMA GEMM): G = A x Fy1^T
    hipLaunchKernelGGL(k_gemm, dim3(1, Mrows / 128), dim3(256), 0, stream,
        bufA, SY_, Fy1, 160, Gbuf, NMODE_, NMODE_, SY_, 5);
    // x-DFT + mode mixing + inverse x-DFT
    hipLaunchKernelGGL(k_spectral_mid, dim3(B_), dim3(256), shb, stream,
        Gbuf, wA[L], wB[L], Hbuf);
    // inverse y (WMMA GEMM): T = H x Fy2^T
    hipLaunchKernelGGL(k_gemm, dim3(2, Mrows / 128), dim3(256), 0, stream,
        Hbuf, NMODE_, Fy2, NMODE_, bufT, SY_, SY_, NMODE_, 1);
    if (L == 0) {
      hipLaunchKernelGGL(k_inorm, dim3(B_ * C_), dim3(256), 0, stream,
          bufT, bufA, (unsigned short*)nullptr, 1);       // h = gelu(inorm(T))
      hipLaunchKernelGGL(k_inorm, dim3(B_ * C_), dim3(256), 0, stream,
          bufA, bufA, (unsigned short*)nullptr, 0);       // inorm for next layer
    } else {
      hipLaunchKernelGGL(k_inorm, dim3(B_ * C_), dim3(256), 0, stream,
          bufT, (float*)nullptr, flath, 1);               // bf16 flat activations
    }
  }

  // hid1 = relu(flat @ out1_w1^T + b1): double-buffered async bf16 WMMA GEMM
  hipLaunchKernelGGL(k_gemm_bf, dim3(22, B_ / 128), dim3(256), 0, stream,
      flath, FLATP_, w1h, FLATP_, hid1h, HID1P_,
      (float*)nullptr, HID1P_, out1_b1, HID1_, HID1KT_, 1);
  // hout = hid1 @ out1_w2^T + b2
  hipLaunchKernelGGL(k_gemm_bf, dim3(2, B_ / 128), dim3(256), 0, stream,
      hid1h, HID1P_, w2h, HID1P_, (unsigned short*)nullptr, 0,
      hout, OUT_, out1_b2, OUT_, HID2KT_, 0);
  // interleave with x0n, reg2 MLP, rescale
  hipLaunchKernelGGL(k_final, dim3(B_), dim3(256), 0, stream,
      hout, x, reg2_w1, reg2_b1, reg2_w2, reg2_b2, outp);

  (void)in_sizes; (void)n_in; (void)out_size; (void)ws_size;
}
